// RecurrentStickBreakingVAE_43662637531302
// MI455X (gfx1250) — matrix operand order repaired and checked by reading, not compiled, not run on hardware
//
#include <hip/hip_runtime.h>
#include <hip/hip_bf16.h>

// ---------------------------------------------------------------------------
// RecurrentStickBreakingVAE forward for MI455X (gfx1250, wave32, WMMA).
//
// Pipeline (all on `stream`, graph-capture safe; no mallocs/syncs):
//   1. cvt x -> bf16 [8192,2048] (K pad), enc_W -> bf16 [2048,1024]
//   2. WMMA GEMM1: hid = relu(x@enc_W+b)            -> bf16 [8192,1024]
//   3. cvt mu_W, lv_W -> bf16
//   4. WMMA GEMM2/3: mu, log_var -> d_out (f32)
//   5. eta_z = mu + eps*exp(lv/2) (hash-based normal, NOT bit-exact with
//      jax.random threefry -- no runtime check available) -> bf16
//   6. persistent 2-LSTM kernel (16 WGs sb / 32 WGs topic), one device-scope
//      atomic barrier per step; c-state lives in registers
//   7. hs^T -> bf16 [512,128] (N pad 99->128), t -> bf16 [128,1024] (K pad)
//   8. WMMA GEMM4: eta = sigmoid(eta_z @ hs^T)      -> f32 [8192,99]
//   9. theta = stick-breaking cumprod               -> bf16 [8192,128]
//  10. WMMA GEMM5: z = theta @ t                    -> bf16 [8192,1024]
//  11. WMMA GEMM6: x_reconst = z@dec_W + b          -> d_out (f32)
// ---------------------------------------------------------------------------

typedef unsigned int   uint;
typedef unsigned short ushort;
typedef unsigned long long u64;
typedef __attribute__((ext_vector_type(16))) __bf16 v16bf;
typedef __attribute__((ext_vector_type(8)))  float  v8f;
typedef __attribute__((ext_vector_type(4)))  uint   v4u;
typedef __attribute__((ext_vector_type(8)))  uint   v8u;

#define Bsz   8192
#define Vv    2000
#define H1    1024
#define LATD  512
#define TOPH  1024
#define LDIM  100

#define XR    (Bsz * Vv)          // x_reconst elements in d_out
#define BL    (Bsz * LATD)        // mu / log_var block

// ---- workspace layout (bytes). Region aliasing relies on stream order. ----
#define OFF0      0u                       // xpad bf16 [8192][2048] (33.5MB)
#define W_ETAZ    (OFF0 + 0u)              // after GEMM1: etaz bf16 [8192][512]
#define W_THETA   (OFF0 + 8388608u)        // theta bf16 [8192][128]
#define W_ZBF     (OFF0 + 10485760u)       // z bf16 [8192][1024]
#define W_ETAF    (OFF0 + 27262976u)       // eta f32 [8192][99]
#define OFF1      33554432u                // encW bf16 [2048][1024] (4.2MB)
#define W_MUW     (OFF1 + 0u)              // after GEMM1: muW bf16 [1024][512]
#define W_LVW     (OFF1 + 1048576u)        // lvW bf16 [1024][512]
#define W_HST     (OFF1 + 2097152u)        // hs^T bf16 [512][128]
#define W_TBF     (OFF1 + 2228224u)        // t bf16 [128][1024]
#define OFF2      37748736u                // hid bf16 [8192][1024] (16.8MB)
#define W_DECW    (OFF2 + 0u)              // after GEMM3: decW bf16 [1024][2048]
#define W_LSTM    (OFF2 + 4194304u)
#define W_HSB     (W_LSTM + 0u)            // h_sb f32 [512]
#define W_HTP     (W_LSTM + 2048u)         // h_tp f32 [1024]
#define W_HSOUT   (W_LSTM + 6144u)         // hs f32 [99][512]
#define W_TOUT    (W_LSTM + 208896u)       // t  f32 [100][1024]
#define W_CTR     (W_LSTM + 618496u)       // int[199] step barrier counters
#define W_LSTM_BYTES 619520u               // zeroed each call
// total workspace need: ~52 MB

__device__ __forceinline__ ushort f2bf(float f) {
  uint u = __float_as_uint(f);
  u += 0x7FFFu + ((u >> 16) & 1u);     // round to nearest even
  return (ushort)(u >> 16);
}

// -------------------- convert + zero-pad f32 -> bf16 ----------------------
// Cpad = 1<<cshift (2048/1024/512/128 all pow2) -> shift/mask indexing.
__global__ void k_cvt_pad(const float* __restrict__ in, ushort* __restrict__ out,
                          int Rreal, int Creal, int cshift) {
  long i = (long)blockIdx.x * blockDim.x + threadIdx.x;
  int r = (int)(i >> cshift), c = (int)(i & ((1 << cshift) - 1));
  float v = (r < Rreal && c < Creal) ? in[(long)r * Creal + c] : 0.0f;
  out[i] = f2bf(v);
}

__global__ void k_zero(int* p, int n) {
  int i = blockIdx.x * blockDim.x + threadIdx.x;
  if (i < n) p[i] = 0;
}

// ---------------------------- WMMA GEMM -----------------------------------
// C[M x Nreal] = epilogue(A[M x Kp](bf16,lda) * B[Kp x Npad](bf16,ldb))
// block 256 thr = 8 waves (4x2), tile 128x128, K step 32, reg-prefetch pipe.
// EP: 0 = bias+relu -> bf16 | 1 = bias -> f32 | 2 = none -> bf16 | 3 = sigmoid -> f32
#define LDSW 40   // ushort stride (80B): keeps b128 alignment, spreads banks

union FragU { uint4 q[2]; v16bf v; };

template <int EP>
__global__ __launch_bounds__(256)
void k_gemm(const ushort* __restrict__ A, const ushort* __restrict__ Bm,
            const float* __restrict__ bias, float* __restrict__ outF,
            ushort* __restrict__ outH, int Kp, int lda, int ldb,
            int Nreal, int ldOut) {
  __shared__ ushort As[128 * LDSW];
  __shared__ ushort Bs[128 * LDSW];
  const int tid  = threadIdx.x;
  const int rowBase = blockIdx.x * 128;
  const int colBase = blockIdx.y * 128;
  const int wave = tid >> 5, lane = tid & 31;
  const int wm = wave & 3, wn = wave >> 2;       // 4 x 2 wave grid
  const int hf = lane >> 4, l16 = lane & 15;

  // load-task geometry
  const int ar0 = tid >> 2, ac8 = tid & 3;       // A: 2 uint4/thread
  const int bkp = tid >> 4, bn8 = tid & 15;      // B: k-pair (2 rows) x 8 cols

  v8f acc[2][4] = {};
  uint4 aq[2], bqa, bqb;

  auto loadG = [&](int kt) {
#pragma unroll
    for (int i = 0; i < 2; ++i)
      aq[i] = *(const uint4*)&A[(long)(rowBase + ar0 + i * 64) * lda + kt + ac8 * 8];
    bqa = *(const uint4*)&Bm[(long)(kt + 2 * bkp) * ldb + colBase + bn8 * 8];
    bqb = *(const uint4*)&Bm[(long)(kt + 2 * bkp + 1) * ldb + colBase + bn8 * 8];
  };

  loadG(0);
  for (int kt = 0; kt < Kp; kt += 32) {
    // ---- commit prefetched tile to LDS ----
#pragma unroll
    for (int i = 0; i < 2; ++i)
      *(uint4*)&As[(ar0 + i * 64) * LDSW + ac8 * 8] = aq[i];
    {
      uint wa[4] = {bqa.x, bqa.y, bqa.z, bqa.w};
      uint wb[4] = {bqb.x, bqb.y, bqb.z, bqb.w};
#pragma unroll
      for (int c = 0; c < 8; ++c) {         // Bs[n][k] transposed, k-pairs packed
        uint lo = (wa[c >> 1] >> ((c & 1) * 16)) & 0xFFFFu;
        uint hi = (wb[c >> 1] >> ((c & 1) * 16)) & 0xFFFFu;
        *(uint*)&Bs[(bn8 * 8 + c) * LDSW + 2 * bkp] = lo | (hi << 16);
      }
    }
    __syncthreads();
    if (kt + 32 < Kp) loadG(kt + 32);       // global loads fly behind the WMMAs

    // ---- fragments per ISA layout (wave32) + 8 WMMAs ----
    FragU af[2];
#pragma unroll
    for (int i = 0; i < 2; ++i) {
      int r0 = wm * 32 + i * 16;
      af[i].q[0] = *(const uint4*)&As[(r0 + l16) * LDSW + hf * 8];
      af[i].q[1] = *(const uint4*)&As[(r0 + l16) * LDSW + 16 + hf * 8];
    }
#pragma unroll
    for (int j = 0; j < 4; ++j) {
      int n0 = wn * 64 + j * 16;
      FragU bf;
      bf.q[0] = *(const uint4*)&Bs[(n0 + l16) * LDSW + hf * 8];
      bf.q[1] = *(const uint4*)&Bs[(n0 + l16) * LDSW + 16 + hf * 8];
#pragma unroll
      for (int i = 0; i < 2; ++i) {
        acc[i][j] = __builtin_amdgcn_wmma_f32_16x16x32_bf16(
            false, af[i].v, false, bf.v, (short)0, acc[i][j], false, false);
      }
    }
    __syncthreads();
  }

  // epilogue: C 16x16 f32 layout: VGPR r -> M = r + 8*hf, N = l16
#pragma unroll
  for (int j = 0; j < 4; ++j) {
    const int col = colBase + wn * 64 + j * 16 + l16;
    if (col < Nreal) {
      const float bv = (EP == 0 || EP == 1) ? bias[col] : 0.0f;
#pragma unroll
      for (int i = 0; i < 2; ++i)
#pragma unroll
        for (int r = 0; r < 8; ++r) {
          const int row = rowBase + wm * 32 + i * 16 + 8 * hf + r;
          float v = acc[i][j][r];
          long o = (long)row * ldOut + col;
          if (EP == 0)      { outH[o] = f2bf(fmaxf(v + bv, 0.0f)); }
          else if (EP == 1) { outF[o] = v + bv; }
          else if (EP == 2) { outH[o] = f2bf(v); }
          else              { outF[o] = 1.0f / (1.0f + expf(-v)); }
        }
    }
  }
}

// ------------------------- eta_z elementwise -------------------------------
__device__ __forceinline__ uint hash32(uint x) {
  x ^= x >> 16; x *= 0x7feb352dU; x ^= x >> 15; x *= 0x846ca68bU; x ^= x >> 16;
  return x;
}
__global__ void k_etaz(const float* __restrict__ dout, ushort* __restrict__ etaz) {
  long i = (long)blockIdx.x * blockDim.x + threadIdx.x;
  if (i >= (long)Bsz * LATD) return;
  float mu = dout[XR + i];
  float lv = dout[XR + BL + i];
  uint h1 = hash32((uint)i * 2u + 0x9E3779B9u);
  uint h2 = hash32((uint)i * 2u + 0x85EBCA6Bu);
  float u1 = ((float)h1 + 0.5f) * 2.3283064365e-10f;
  float u2 = ((float)h2 + 0.5f) * 2.3283064365e-10f;
  float eps = sqrtf(-2.0f * logf(u1)) * cosf(6.28318530718f * u2);
  etaz[i] = f2bf(fmaf(eps, expf(0.5f * lv), mu));
}

// ------------------- persistent dual-LSTM (zero input) ---------------------
// Block b owns hidden slice [lb*32, lb*32+32); its 128 threads compute all 4
// gates for the slice (gate = tid>>5, torch order i,f,g,o), so the c/h update
// is block-local (c in registers). One device-scope counter barrier per step.
__device__ __forceinline__ float sigm(float x) { return 1.0f / (1.0f + expf(-x)); }

__global__ __launch_bounds__(128)
void k_lstm(const float* __restrict__ WhhS, const float* __restrict__ bS,
            float* hS, float* hsS, int* ctrS, int nbS,
            const float* __restrict__ WhhT, const float* __restrict__ bT,
            float* hT, float* hsT, int* ctrT, int nbT) {
  const int bid = blockIdx.x;
  const float *Whh, *bb; float *h, *hso; int *ctr; int H, T, nb, lb;
  if (bid < nbS) { Whh=WhhS; bb=bS; h=hS; hso=hsS; ctr=ctrS; H=LATD; T=LDIM-1; nb=nbS; lb=bid; }
  else           { Whh=WhhT; bb=bT; h=hT; hso=hsT; ctr=ctrT; H=TOPH; T=LDIM;   nb=nbT; lb=bid-nbS; }

  __shared__ float hsh[TOPH];
  __shared__ float gsh[128];
  const int tid = threadIdx.x;
  const int hL = tid & 31, gate = tid >> 5;
  const int hIdx = lb * 32 + hL;
  const long row = (long)gate * H + hIdx;
  const float4* w4 = (const float4*)(Whh + row * H);
  const float bias = bb[row];
  float c = 0.0f;

  for (int t = 0; t < T; ++t) {
    for (int k = tid; k < H; k += 128) hsh[k] = h[k];   // h==0 at t=0 (pre-zeroed)
    __syncthreads();
    float g = bias;
    const float4* h4 = (const float4*)hsh;
    const int n4 = H >> 2;
    for (int k = 0; k < n4; ++k) {
      float4 a = w4[k], b = h4[k];
      g = fmaf(a.x, b.x, g); g = fmaf(a.y, b.y, g);
      g = fmaf(a.z, b.z, g); g = fmaf(a.w, b.w, g);
    }
    gsh[tid] = g;
    __syncthreads();
    if (gate == 0) {
      float gi = sigm(gsh[hL]);
      float gf = sigm(gsh[32 + hL]);
      float gg = tanhf(gsh[64 + hL]);
      float go = sigm(gsh[96 + hL]);
      c = fmaf(gf, c, gi * gg);
      float hn = go * tanhf(c);
      h[hIdx] = hn;
      hso[(long)t * H + hIdx] = hn;
    }
    __threadfence();                                    // publish h to L2
    if (tid == 0) {
      __hip_atomic_fetch_add(&ctr[t], 1, __ATOMIC_RELEASE, __HIP_MEMORY_SCOPE_AGENT);
      while (__hip_atomic_load(&ctr[t], __ATOMIC_ACQUIRE, __HIP_MEMORY_SCOPE_AGENT) < nb)
        __builtin_amdgcn_s_sleep(2);
    }
    __syncthreads();                                    // broadcast barrier pass
  }
}

// hs [99][512] f32 -> hs^T bf16 [512][128] (cols >= 99 zero)
__global__ void k_hsT(const float* __restrict__ hs, ushort* __restrict__ out) {
  int i = blockIdx.x * blockDim.x + threadIdx.x;
  if (i >= 512 * 128) return;
  int k = i >> 7, n = i & 127;
  out[i] = (n < LDIM - 1) ? f2bf(hs[(long)n * 512 + k]) : (ushort)0;
}

// stick-breaking: theta[b,l] = eta[b,l] * prod_{j<l}(1-eta[b,j]); eta[:,99]=1
__global__ void k_theta(const float* __restrict__ eta, ushort* __restrict__ theta) {
  int b = blockIdx.x * blockDim.x + threadIdx.x;
  if (b >= Bsz) return;
  const float* er = eta + (long)b * (LDIM - 1);
  ushort* tr = theta + (long)b * 128;
  float prefix = 1.0f;
  for (int l = 0; l < LDIM; ++l) {
    float e = (l < LDIM - 1) ? er[l] : 1.0f;
    tr[l] = f2bf(e * prefix);
    prefix *= (1.0f - e);
  }
  for (int l = LDIM; l < 128; ++l) tr[l] = 0;
}

// ------------------ TDM / cluster compile-probe (NOT launched) -------------
// Exercises the CDNA5 async-tensor path: builds a Tensor DMA Descriptor (D#)
// in SGPR groups and issues tensor_load_to_lds via inline asm (portable
// across ROCm7.2 / clang-23 builtin-arity differences), then waits on
// TENSORcnt. Kept off the execution path: D# bit packing is from the
// PDF-extracted ISA tables and cannot be validated without hardware.
__global__ void k_tdm_probe(const float* __restrict__ src, float* __restrict__ dst) {
  __shared__ float tile[64 * 64];
  union { u64 d[2]; v4u v; } g0;
  union { u64 d[4]; v8u v; } g1;
  u64 ga = (u64)(size_t)src;
  uint ldsOff = (uint)(size_t)(void*)tile;            // addr[31:0] == LDS offset
  g0.d[0] = 1ull | ((u64)ldsOff << 32);               // count=1 | lds_addr
  g0.d[1] = (ga & 0x01FFFFFFFFFFFFFFull) | (2ull << 62);  // global_addr | type=2
  g1.d[0] = (2ull << 16) | (64ull << 48);             // data_size=4B, tensor_dim0=64
  g1.d[1] = (64ull << 16) | (64ull << 48);            // tensor_dim1=64, tile_dim0=64
  g1.d[2] = 64ull | (64ull << 32);                    // tile_dim1=64, dim0_stride=64
  g1.d[3] = 0ull;                                     // dim1_stride=0 (2D)
  v4u gz = {0u, 0u, 0u, 0u};
  if (threadIdx.x == 0 && __builtin_amdgcn_cluster_id_x() == 0) {
    asm volatile("tensor_load_to_lds %0, %1, %2, %3"
                 :: "s"(g0.v), "s"(g1.v), "s"(gz), "s"(gz) : "memory");
  }
  __builtin_amdgcn_s_wait_tensorcnt(0);
  __builtin_amdgcn_s_cluster_barrier();               // NOP when not in cluster
  __syncthreads();
  dst[blockIdx.x * blockDim.x + threadIdx.x] = tile[threadIdx.x & 4095];
}

// ---------------------------------------------------------------------------
extern "C" void kernel_launch(void* const* d_in, const int* in_sizes, int n_in,
                              void* d_out, int out_size, void* d_ws, size_t ws_size,
                              hipStream_t stream) {
  const float* x     = (const float*)d_in[0];
  const float* encW  = (const float*)d_in[2];
  const float* encB  = (const float*)d_in[3];
  const float* muW   = (const float*)d_in[4];
  const float* muB   = (const float*)d_in[5];
  const float* lvW   = (const float*)d_in[6];
  const float* lvB   = (const float*)d_in[7];
  const float* whhSb = (const float*)d_in[8];
  const float* bSb   = (const float*)d_in[9];
  const float* whhTp = (const float*)d_in[10];
  const float* bTp   = (const float*)d_in[11];
  const float* decW  = (const float*)d_in[12];
  const float* decB  = (const float*)d_in[13];
  float* outF = (float*)d_out;
  char*  ws   = (char*)d_ws;

  ushort* xpad = (ushort*)(ws + OFF0);
  ushort* ewB  = (ushort*)(ws + OFF1);
  ushort* hid  = (ushort*)(ws + OFF2);
  ushort* muWb = (ushort*)(ws + W_MUW);
  ushort* lvWb = (ushort*)(ws + W_LVW);
  ushort* etaz = (ushort*)(ws + W_ETAZ);
  ushort* hstB = (ushort*)(ws + W_HST);
  ushort* tBf  = (ushort*)(ws + W_TBF);
  ushort* thB  = (ushort*)(ws + W_THETA);
  ushort* zBf  = (ushort*)(ws + W_ZBF);
  ushort* dWb  = (ushort*)(ws + W_DECW);
  float*  etaF = (float*)(ws + W_ETAF);
  float*  hSb  = (float*)(ws + W_HSB);
  float*  hTp  = (float*)(ws + W_HTP);
  float*  hsO  = (float*)(ws + W_HSOUT);
  float*  tO   = (float*)(ws + W_TOUT);
  int*    ctr  = (int*)(ws + W_CTR);

  auto cdiv = [](long a, long b) { return (int)((a + b - 1) / b); };

  // 1. pack x and enc_W (K: 2000 -> 2048)
  k_cvt_pad<<<cdiv((long)Bsz * 2048, 256), 256, 0, stream>>>(x, xpad, Bsz, Vv, 11);
  k_cvt_pad<<<cdiv(2048L * H1, 256), 256, 0, stream>>>(encW, ewB, Vv, H1, 10);
  // 2. hid = relu(x@enc_W + b)  (bf16 out)
  k_gemm<0><<<dim3(Bsz / 128, H1 / 128), 256, 0, stream>>>(
      xpad, ewB, encB, nullptr, hid, 2048, 2048, H1, H1, H1);
  // 3. pack mu_W / lv_W (alias over enc_W region -- enc_W now dead)
  k_cvt_pad<<<cdiv((long)H1 * LATD, 256), 256, 0, stream>>>(muW, muWb, H1, LATD, 9);
  k_cvt_pad<<<cdiv((long)H1 * LATD, 256), 256, 0, stream>>>(lvW, lvWb, H1, LATD, 9);
  // 4. mu, log_var -> d_out (f32)
  k_gemm<1><<<dim3(Bsz / 128, LATD / 128), 256, 0, stream>>>(
      hid, muWb, muB, outF + XR, nullptr, H1, H1, LATD, LATD, LATD);
  k_gemm<1><<<dim3(Bsz / 128, LATD / 128), 256, 0, stream>>>(
      hid, lvWb, lvB, outF + XR + BL, nullptr, H1, H1, LATD, LATD, LATD);
  // 5. eta_z (aliases xpad region -- xpad dead after GEMM1)
  k_etaz<<<cdiv((long)Bsz * LATD, 256), 256, 0, stream>>>(outF, etaz);
  // 6. zero LSTM state/counters (aliases hid region -- hid dead), pack dec_W
  k_zero<<<cdiv(W_LSTM_BYTES / 4, 256), 256, 0, stream>>>((int*)(ws + W_LSTM), W_LSTM_BYTES / 4);
  k_cvt_pad<<<cdiv((long)H1 * 2048, 256), 256, 0, stream>>>(decW, dWb, H1, Vv, 11);
  // 7. both LSTMs, persistent (16 + 32 blocks, all co-resident)
  k_lstm<<<48, 128, 0, stream>>>(whhSb, bSb, hSb, hsO, ctr, 16,
                                 whhTp, bTp, hTp, tO, ctr + (LDIM - 1), 32);
  // 8. pack hs^T (N pad 99->128) and t (K pad 100->128)
  k_hsT<<<cdiv(512L * 128, 256), 256, 0, stream>>>(hsO, hstB);
  k_cvt_pad<<<cdiv(128L * TOPH, 256), 256, 0, stream>>>(tO, tBf, LDIM, TOPH, 10);
  // 9. eta = sigmoid(eta_z @ hs^T)  -> f32 [8192,99]
  k_gemm<3><<<dim3(Bsz / 128, 1), 256, 0, stream>>>(
      etaz, hstB, nullptr, etaF, nullptr, LATD, LATD, 128, LDIM - 1, LDIM - 1);
  // 10. theta (bf16, K-padded to 128)
  k_theta<<<cdiv(Bsz, 128), 128, 0, stream>>>(etaF, thB);
  // 11. z = theta @ t  (bf16 out)
  k_gemm<2><<<dim3(Bsz / 128, TOPH / 128), 256, 0, stream>>>(
      thB, tBf, nullptr, nullptr, zBf, 128, 128, TOPH, TOPH, TOPH);
  // 12. x_reconst = z @ dec_W + b -> d_out (f32, N guard 2000)
  k_gemm<1><<<dim3(Bsz / 128, 2048 / 128), 256, 0, stream>>>(
      zBf, dWb, decB, outF, nullptr, H1, H1, 2048, Vv, Vv);
}
